// SeqLabelModel_82652350644479
// MI455X (gfx1250) — matrix operand order repaired and checked by reading, not compiled, not run on hardware
//
#include <hip/hip_runtime.h>
#include <hip/hip_bf16.h>
#include <stdint.h>

#define B_ 64
#define T_ 256
#define E_ 256
#define H_ 512
#define KT 4
#define START_ 2
#define STOP_ 3
#define NEGV -10000.0f
#define NWG 8

typedef __attribute__((ext_vector_type(16))) __bf16 v16bf;
typedef __attribute__((ext_vector_type(8)))  float  v8f;
typedef __hip_bfloat16 bf16_t;

union Frag16 { v16bf v; uint4 q[2]; };

__device__ __forceinline__ v8f wmma_bf16(v16bf a, v16bf b, v8f c) {
  // D = A(16x32 bf16) * B(32x16 bf16) + C(f32), wave32
  return __builtin_amdgcn_wmma_f32_16x16x32_bf16(false, a, false, b, (short)0, c, false, false);
}

__device__ __forceinline__ float sigm(float x) { return 1.0f / (1.0f + __expf(-x)); }

// A fragment (16x32 bf16, MxK), row-major source, ld in elements.
// lanes 0-15: M=lane, K = {0..7, 16..23}; lanes 16-31: M=lane-16, K = {8..15, 24..31}
__device__ __forceinline__ v16bf load_a(const bf16_t* __restrict__ base, int mbase, int ld,
                                        int kb, int lane) {
  int half = lane >> 4, m = lane & 15;
  const bf16_t* p = base + (size_t)(mbase + m) * ld + kb + half * 8;
  Frag16 f;
  f.q[0] = *(const uint4*)(p);       // K = kb + half*8 + 0..7
  f.q[1] = *(const uint4*)(p + 16);  // K = kb + 16 + half*8 + 0..7
  return f.v;
}

// B fragment (32x16 bf16, KxN). B column n is row n of W (row-major).
// lanes 0-15: N=lane, K = {0..7, 16..23}; lanes 16-31: N=lane-16, K = {8..15, 24..31}
__device__ __forceinline__ v16bf load_b(const bf16_t* __restrict__ rowptr, int kb, int lane) {
  int half = lane >> 4;
  const bf16_t* p = rowptr + kb + half * 8;
  Frag16 f;
  f.q[0] = *(const uint4*)(p);
  f.q[1] = *(const uint4*)(p + 16);
  return f.v;
}

__device__ __forceinline__ void gridbar(unsigned* cnt, unsigned* gen, unsigned nwg) {
  __syncthreads();
  if (threadIdx.x == 0) {
    __threadfence();
    unsigned g = __hip_atomic_load(gen, __ATOMIC_RELAXED, __HIP_MEMORY_SCOPE_AGENT);
    unsigned a = __hip_atomic_fetch_add(cnt, 1u, __ATOMIC_ACQ_REL, __HIP_MEMORY_SCOPE_AGENT);
    if (a == nwg - 1) {
      __hip_atomic_store(cnt, 0u, __ATOMIC_RELAXED, __HIP_MEMORY_SCOPE_AGENT);
      __hip_atomic_fetch_add(gen, 1u, __ATOMIC_ACQ_REL, __HIP_MEMORY_SCOPE_AGENT);
    } else {
      while (__hip_atomic_load(gen, __ATOMIC_ACQUIRE, __HIP_MEMORY_SCOPE_AGENT) == g)
        __builtin_amdgcn_s_sleep(2);
    }
    __threadfence();
  }
  __syncthreads();
}

// ---------------- embedding gather + f32 -> bf16 ----------------
__global__ void k_embed(const int* __restrict__ sent, const float* __restrict__ emb,
                        bf16_t* __restrict__ xbf) {
  int t = blockIdx.x, b = blockIdx.y;
  int tok = sent[b * T_ + t];
  const float* src = emb + (size_t)tok * E_;
  bf16_t* dst = xbf + ((size_t)t * B_ + b) * E_;
  for (int i = threadIdx.x; i < E_; i += blockDim.x) dst[i] = __float2bfloat16(src[i]);
}

__global__ void k_cvt(const float* __restrict__ src, bf16_t* __restrict__ dst, int n) {
  int i = blockIdx.x * blockDim.x + threadIdx.x;
  if (i < n) dst[i] = __float2bfloat16(src[i]);
}

__global__ void k_bias(const float* __restrict__ a, const float* __restrict__ b,
                       float* __restrict__ dst, int n) {
  int i = blockIdx.x * blockDim.x + threadIdx.x;
  if (i < n) dst[i] = a[i] + b[i];
}

// h0 -> bf16 double-buffer slot 0 for all 4 (layer,dir) units; zero barriers
__global__ void k_init(const float* __restrict__ h0, bf16_t* __restrict__ hbuf,
                       unsigned* __restrict__ bar) {
  int i = blockIdx.x * blockDim.x + threadIdx.x;
  const int n = 4 * B_ * H_;
  if (i < n) {
    int u = i / (B_ * H_);
    int off = i % (B_ * H_);
    hbuf[((size_t)u * 2) * B_ * H_ + off] = __float2bfloat16(h0[i]);
  }
  if (i < 16) bar[i] = 0;
}

// ---------------- persistent bidirectional LSTM layer ----------------
// grid: (NWG, 2). blockIdx.y = direction (0 fwd, 1 bwd). 512 threads = 16 waves.
// WG owns 64 h-columns; wave w: N-tile = w&3 (16 cols), M-tile = w>>2 (16 batch rows).
// Per-wave tile: 4 gates x 1 M-tile -> 4 accumulators (32 VGPRs), no spills.
template <int DIN>
__global__ void __launch_bounds__(512, 1) k_lstm(
    const bf16_t* __restrict__ xin,                 // [T,B,DIN] bf16
    const bf16_t* __restrict__ wihF, const bf16_t* __restrict__ whhF, const float* __restrict__ biasF,
    const bf16_t* __restrict__ wihB, const bf16_t* __restrict__ whhB, const float* __restrict__ biasB,
    const float* __restrict__ c0,                   // [4,B,H] f32
    bf16_t* __restrict__ hbuf,                      // [4 units][2 bufs][B,H] bf16
    bf16_t* __restrict__ xout,                      // [T,B,2H] bf16
    unsigned* __restrict__ bar, int layer) {
  const int dir = blockIdx.y;
  const int unit = layer * 2 + dir;
  const bf16_t* wih = dir ? wihB : wihF;
  const bf16_t* whh = dir ? whhB : whhF;
  const float* bias = dir ? biasB : biasF;
  const size_t BH = (size_t)B_ * H_;
  bf16_t* hbase = hbuf + (size_t)(unit * 2) * BH;  // double buffer: [2][B,H]
  unsigned* cnt = bar + (unit * 2);
  unsigned* gen = bar + (unit * 2) + 1;

  const int g = blockIdx.x;                 // column-slice owner
  const int w = threadIdx.x >> 5;
  const int lane = threadIdx.x & 31;
  const int half = lane >> 4;
  const int ntile = w & 3;
  const int mtile = w >> 2;
  const int mbase = mtile * 16;             // M-tile (batch rows)
  const int col = g * 64 + ntile * 16 + (lane & 15);  // owned h column (0..511)

  // biases per gate (gate order: i,f,g,o blocks of H)
  const float bI = bias[0 * H_ + col];
  const float bF = bias[1 * H_ + col];
  const float bG = bias[2 * H_ + col];
  const float bO = bias[3 * H_ + col];

  // per-gate B row base pointers (column 'col' of B == row of W)
  const bf16_t* wihr[4];
  const bf16_t* whhr[4];
#pragma unroll
  for (int gi = 0; gi < 4; ++gi) {
    wihr[gi] = wih + (size_t)(gi * H_ + col) * DIN;
    whhr[gi] = whh + (size_t)(gi * H_ + col) * H_;
  }

  // cell state lives in registers, mapped exactly like the D fragment
  float creg[8];
  const float* c0u = c0 + (size_t)unit * BH;
#pragma unroll
  for (int r = 0; r < 8; ++r) {
    int row = mbase + r + 8 * half;
    creg[r] = c0u[(size_t)row * H_ + col];
  }

  const v8f zf = {0.f, 0.f, 0.f, 0.f, 0.f, 0.f, 0.f, 0.f};

  for (int s = 0; s < T_; ++s) {
    const int t = dir ? (T_ - 1 - s) : s;
    const bf16_t* xrow = xin + (size_t)t * B_ * DIN;
    // double-buffer select via integer offset so accesses stay in global addrspace
    const size_t roff = (s & 1) ? BH : 0;
    const bf16_t* hread = hbase + roff;
    bf16_t* hwrite = hbase + (BH - roff);

    if (s + 1 < T_) {
      const int tn = dir ? (T_ - 2 - s) : (s + 1);
      __builtin_prefetch(xin + ((size_t)tn * B_ + (mbase + (lane & 15))) * DIN, 0, 3);
    }

    v8f acc[4];
#pragma unroll
    for (int gi = 0; gi < 4; ++gi) acc[gi] = zf;

    // x_t @ W_ih.T contribution
#pragma unroll 2
    for (int kb = 0; kb < DIN; kb += 32) {
      v16bf a = load_a(xrow, mbase, DIN, kb, lane);
#pragma unroll
      for (int gi = 0; gi < 4; ++gi) {
        v16bf bfr = load_b(wihr[gi], kb, lane);
        acc[gi] = wmma_bf16(a, bfr, acc[gi]);
      }
    }
    // h_{t-1} @ W_hh.T contribution
#pragma unroll 2
    for (int kb = 0; kb < H_; kb += 32) {
      v16bf a = load_a(hread, mbase, H_, kb, lane);
#pragma unroll
      for (int gi = 0; gi < 4; ++gi) {
        v16bf bfr = load_b(whhr[gi], kb, lane);
        acc[gi] = wmma_bf16(a, bfr, acc[gi]);
      }
    }

    // LSTM cell epilogue, elementwise on the D fragment
#pragma unroll
    for (int r = 0; r < 8; ++r) {
      float iv = sigm(acc[0][r] + bI);
      float fv = sigm(acc[1][r] + bF);
      float gv = tanhf(acc[2][r] + bG);
      float ov = sigm(acc[3][r] + bO);
      float c = fv * creg[r] + iv * gv;
      creg[r] = c;
      float h = ov * tanhf(c);
      int row = mbase + r + 8 * half;
      bf16_t hx = __float2bfloat16(h);
      hwrite[(size_t)row * H_ + col] = hx;
      xout[(size_t)t * B_ * (2 * H_) + (size_t)row * (2 * H_) + dir * H_ + col] = hx;
    }

    gridbar(cnt, gen, NWG);
  }
}

// ---------------- output projection: feats = out @ w_out.T + b_out (K=4) ----------------
__global__ void k_proj(const bf16_t* __restrict__ outb, const float* __restrict__ w_out,
                       const float* __restrict__ b_out, float* __restrict__ feats) {
  int row = blockIdx.x;  // t*B + b
  int tid = threadIdx.x; // 128
  const bf16_t* xr = outb + (size_t)row * 1024;
  float p0 = 0.f, p1 = 0.f, p2 = 0.f, p3 = 0.f;
  for (int i = tid; i < 1024; i += 128) {
    float x = __bfloat162float(xr[i]);
    p0 += x * w_out[i];
    p1 += x * w_out[1024 + i];
    p2 += x * w_out[2048 + i];
    p3 += x * w_out[3072 + i];
  }
  __shared__ float red[4][128];
  red[0][tid] = p0; red[1][tid] = p1; red[2][tid] = p2; red[3][tid] = p3;
  __syncthreads();
  for (int sft = 64; sft > 0; sft >>= 1) {
    if (tid < sft) {
#pragma unroll
      for (int k = 0; k < 4; ++k) red[k][tid] += red[k][tid + sft];
    }
    __syncthreads();
  }
  if (tid < 4) feats[(size_t)row * 4 + tid] = red[tid][0] + b_out[tid];
}

// ---------------- Viterbi decode, K=4, backpointers packed 2 bits each ----------------
__global__ void k_viterbi(const float* __restrict__ feats, const float* __restrict__ trans,
                          uint8_t* __restrict__ bp, float* __restrict__ out) {
  int b = threadIdx.x;
  if (b >= B_) return;
  float tr[4][4];
#pragma unroll
  for (int n = 0; n < 4; ++n)
#pragma unroll
    for (int p = 0; p < 4; ++p) tr[n][p] = trans[n * 4 + p];

  float fv[4];
#pragma unroll
  for (int k = 0; k < 4; ++k) fv[k] = (k == START_) ? 0.f : NEGV;

  for (int t = 0; t < T_; ++t) {
    const float* ft = feats + ((size_t)t * B_ + b) * 4;
    float nf[4];
    unsigned byte = 0;
#pragma unroll
    for (int n = 0; n < 4; ++n) {
      float best = fv[0] + tr[n][0];
      int arg = 0;
#pragma unroll
      for (int p = 1; p < 4; ++p) {
        float sc = fv[p] + tr[n][p];
        if (sc > best) { best = sc; arg = p; }
      }
      nf[n] = best + ft[n];
      byte |= (unsigned)arg << (2 * n);
    }
    bp[(size_t)t * B_ + b] = (uint8_t)byte;
#pragma unroll
    for (int k = 0; k < 4; ++k) fv[k] = nf[k];
  }

  float best = fv[0] + tr[STOP_][0];
  int arg = 0;
#pragma unroll
  for (int k = 1; k < 4; ++k) {
    float sc = fv[k] + tr[STOP_][k];
    if (sc > best) { best = sc; arg = k; }
  }
  out[b] = best;
  int tag = arg;
  for (int t = T_ - 1; t >= 0; --t) {
    out[B_ + (size_t)b * T_ + t] = (float)tag;
    tag = (bp[(size_t)t * B_ + b] >> (2 * tag)) & 3;
  }
}

extern "C" void kernel_launch(void* const* d_in, const int* in_sizes, int n_in,
                              void* d_out, int out_size, void* d_ws, size_t ws_size,
                              hipStream_t stream) {
  (void)in_sizes; (void)n_in; (void)out_size; (void)ws_size;
  const int*   sent = (const int*)d_in[0];
  const float* emb  = (const float*)d_in[1];
  // unit order: l0f, l0b, l1f, l1b
  const float* wih_f32[4] = {(const float*)d_in[2], (const float*)d_in[6],
                             (const float*)d_in[10], (const float*)d_in[14]};
  const float* whh_f32[4] = {(const float*)d_in[3], (const float*)d_in[7],
                             (const float*)d_in[11], (const float*)d_in[15]};
  const float* bih[4] = {(const float*)d_in[4], (const float*)d_in[8],
                         (const float*)d_in[12], (const float*)d_in[16]};
  const float* bhh[4] = {(const float*)d_in[5], (const float*)d_in[9],
                         (const float*)d_in[13], (const float*)d_in[17]};
  const float* h0 = (const float*)d_in[18];
  const float* c0 = (const float*)d_in[19];
  const float* w_out = (const float*)d_in[20];
  const float* b_out = (const float*)d_in[21];
  const float* trans = (const float*)d_in[22];
  float* out = (float*)d_out;

  char* ws = (char*)d_ws;
  size_t off = 0;
  auto alloc = [&](size_t bytes) -> char* {
    char* p = ws + off;
    off = (off + bytes + 255) & ~(size_t)255;
    return p;
  };
  bf16_t* xbf  = (bf16_t*)alloc((size_t)T_ * B_ * E_ * 2);
  bf16_t* x1   = (bf16_t*)alloc((size_t)T_ * B_ * 1024 * 2);
  bf16_t* outb = (bf16_t*)alloc((size_t)T_ * B_ * 1024 * 2);
  const int wihN[4] = {2048 * 256, 2048 * 256, 2048 * 1024, 2048 * 1024};
  bf16_t* wihb[4]; bf16_t* whhb[4]; float* biasb[4];
  for (int u = 0; u < 4; ++u) {
    wihb[u]  = (bf16_t*)alloc((size_t)wihN[u] * 2);
    whhb[u]  = (bf16_t*)alloc((size_t)2048 * 512 * 2);
    biasb[u] = (float*)alloc((size_t)2048 * 4);
  }
  bf16_t*   hbuf  = (bf16_t*)alloc((size_t)4 * 2 * B_ * H_ * 2);
  float*    feats = (float*)alloc((size_t)T_ * B_ * KT * 4);
  uint8_t*  bp    = (uint8_t*)alloc((size_t)T_ * B_);
  unsigned* bar   = (unsigned*)alloc(64);

  // 1) embedding gather -> bf16, time-major [T,B,E]
  k_embed<<<dim3(T_, B_), 64, 0, stream>>>(sent, emb, xbf);
  // 2) weights -> bf16 (L2-resident working set), biases combined
  for (int u = 0; u < 4; ++u) {
    k_cvt<<<(wihN[u] + 255) / 256, 256, 0, stream>>>(wih_f32[u], wihb[u], wihN[u]);
    k_cvt<<<(2048 * 512 + 255) / 256, 256, 0, stream>>>(whh_f32[u], whhb[u], 2048 * 512);
    k_bias<<<8, 256, 0, stream>>>(bih[u], bhh[u], biasb[u], 2048);
  }
  // 3) h0 -> bf16 double buffers, zero barriers
  k_init<<<(4 * B_ * H_ + 255) / 256, 256, 0, stream>>>(h0, hbuf, bar);
  // 4) layer 0 (fwd+bwd persistent), writes x1 = concat(hf, hb)
  k_lstm<E_><<<dim3(NWG, 2), 512, 0, stream>>>(
      xbf, wihb[0], whhb[0], biasb[0], wihb[1], whhb[1], biasb[1],
      c0, hbuf, x1, bar, 0);
  // 5) layer 1, writes outb = concat(hf1, hb1)
  k_lstm<2 * H_><<<dim3(NWG, 2), 512, 0, stream>>>(
      x1, wihb[2], whhb[2], biasb[2], wihb[3], whhb[3], biasb[3],
      c0, hbuf, outb, bar, 1);
  // 6) projection to K=4 feats
  k_proj<<<T_ * B_, 128, 0, stream>>>(outb, w_out, b_out, feats);
  // 7) Viterbi decode -> d_out = [scores(B), path(B*T)]
  k_viterbi<<<1, 64, 0, stream>>>(feats, trans, bp, out);
}
